// KANLayer_9165460210146
// MI455X (gfx1250) — matrix operand order repaired and checked by reading, not compiled, not run on hardware
//
#include <hip/hip_runtime.h>

// ---------------------------------------------------------------------------
// KAN layer as on-the-fly GEMM:  out[8192x16] = basis[8192x36864] * W[36864x16]
//   basis[n,(g,d)] = 1 - |x[n,d] - grid[g]|   (relu redundant: |x-g| <= 1)
// WMMA f16 / f32 accumulate.  B fragments staged to LDS with the Tensor Data
// Mover (double-buffered, TENSORcnt-synchronized).  A and B fragments are
// software-pipelined; K split 6-way, deterministic fixed-order reduction.
// ---------------------------------------------------------------------------

typedef _Float16     v16h  __attribute__((ext_vector_type(16)));
typedef float        v8f   __attribute__((ext_vector_type(8)));
typedef unsigned int u32x4 __attribute__((ext_vector_type(4)));
typedef int          i32x4 __attribute__((ext_vector_type(4)));
typedef int          i32x8 __attribute__((ext_vector_type(8)));

#define KB     8192          // batch (M)
#define KD     192           // grid size == input dim
#define KO     16            // outputs (N)
#define NCHUNK (6 * 192)     // K chunks of 32  (K = 192*192 = 36864)
#define CHUNK_HALF 512       // 32 lanes * 16 halves per chunk (1 KiB)
#define PART_ELEMS (KB * KO) // one partial output buffer
#define STAGE_CHUNKS 16      // chunks staged per barrier round (16 KiB)
#define NSTAGE 12            // 192 chunks / 16

// ---------------------------------------------------------------------------
// Prep: permute splines[O=16, G=192, D=192] (f32) into per-chunk B fragments
// (f16) in 32x16 B-fragment lane order (contiguous k, same order as the A side):
//   lane = 16*group + N ;  half h -> K = h + 16*group ;  d = db*32 + K
// ---------------------------------------------------------------------------
__global__ __launch_bounds__(256) void kan_pack_splines(
    const float* __restrict__ splines, _Float16* __restrict__ bpack) {
  const int idx   = blockIdx.x * 256 + threadIdx.x;   // 0 .. 589823 (exact)
  const int h     = idx & 15;
  const int lane  = (idx >> 4) & 31;
  const int chunk = idx >> 9;                         // 0 .. 1151
  const int db    = chunk / 192;                      // d-block 0..5
  const int g     = chunk - db * 192;                 // 0..191
  const int N     = lane & 15;
  const int grp   = lane >> 4;
  const int K     = h + 16 * grp;
  const int d     = db * 32 + K;
  bpack[idx] = (_Float16)splines[((size_t)N * KD + g) * KD + d];
}

// ---------------------------------------------------------------------------
// TDM: DMA 4 KiB (1024 dwords) of B fragments from global into LDS.
// D# group0: count=1, lds_addr, global_addr[56:0], type=2 ("image").
// D# group1: data_size=4B, tensor_dim0=1024, tensor_dim1=1, tile_dim0=1024,
//            tile_dim1=1, tensor_dim0_stride=1024.
// ---------------------------------------------------------------------------
__device__ __forceinline__ void tdm_load_4k(unsigned lds_byte_off,
                                            unsigned long long gaddr) {
  u32x4 g0;
  g0.x = 1u;                                        // count=1, user D#
  g0.y = lds_byte_off;                              // lds_addr [63:32]
  g0.z = (unsigned)gaddr;                           // global_addr[31:0]
  g0.w = (unsigned)(gaddr >> 32) | 0x80000000u;     // addr[56:32] | type=2<<30
  i32x8 g1;
  g1[0] = 0x00020000;   // workgroup_mask=0, data_size=2 (4 bytes)
  g1[1] = 1024 << 16;   // tensor_dim0[15:0]=1024 (atomic_barrier_addr=0)
  g1[2] = 1 << 16;      // tensor_dim0[31:16]=0 | tensor_dim1[15:0]=1
  g1[3] = 1024 << 16;   // tensor_dim1[31:16]=0 | tile_dim0=1024
  g1[4] = 1;            // tile_dim1=1, tile_dim2=0
  g1[5] = 1024;         // tensor_dim0_stride[31:0]=1024
  g1[6] = 0;
  g1[7] = 0;
  i32x4 z4 = {0, 0, 0, 0};
  i32x8 z8 = {0, 0, 0, 0, 0, 0, 0, 0};
  // 6-arg form (clang-23 / therock-10.0 headers): groups 0..3 + extra + cpol
  __builtin_amdgcn_tensor_load_to_lds(g0, g1, z4, z4, z8, 0);
}

// basis fragment: 1 - |xp - grid[g]|   (3 packed-f16 VALU ops per dword)
__device__ __forceinline__ v16h basisv(v16h xp, float gval) {
  const _Float16 gh = (_Float16)gval;
  return (_Float16)1.0f - __builtin_elementwise_abs(xp - gh);
}

// ---------------------------------------------------------------------------
// GEMM: block = 4 waves sharing one K-split (kv) over 4 adjacent M tiles.
// B fragments double-buffered in LDS via TDM (2 x 16 KiB); each wave DMAs its
// 4 KiB quarter of the next stage while consuming the current one from LDS.
// A fragment ping-ponged (a_cur/a_nxt), B pipelined at distance 2.
// A/B fragment k order here (contiguous): lanes0-15 K={0..15}, lanes16-31 K={16..31}
// ---------------------------------------------------------------------------
__global__ __launch_bounds__(128) void kan_wmma_gemm(
    const float* __restrict__ x, const float* __restrict__ grid,
    const _Float16* __restrict__ bpack, float* __restrict__ part) {
  __shared__ _Float16 smem[2 * STAGE_CHUNKS * CHUNK_HALF];  // 32 KiB

  const int lane  = threadIdx.x & 31;
  const int wid   = __builtin_amdgcn_readfirstlane(threadIdx.x >> 5); // SGPR
  const int kv    = blockIdx.x % 6;        // K split (d-block), wave-uniform
  const int tg    = blockIdx.x / 6;        // tile group 0..127
  const int tile  = tg * 4 + wid;          // 512 M tiles of 16 rows
  const int n0    = tile * 16;
  const int grp   = lane >> 4;
  const int row   = n0 + (lane & 15);

  // Pack this wave's x slice into A-fragment order (once per wave).
  // K(h) = h + 16*grp  ->  xr pre-offset by 16*grp.
  const float* xr = x + (size_t)row * KD + kv * 32 + grp * 16;
  v16h xp;
#pragma unroll
  for (int h = 0; h < 8; ++h)  xp[h] = (_Float16)xr[h];
#pragma unroll
  for (int h = 8; h < 16; ++h) xp[h] = (_Float16)xr[h];

  const unsigned lds_base =
      (unsigned)(unsigned long long)(&smem[0]);
  const unsigned long long gbase =
      (unsigned long long)(bpack) + (unsigned long long)(kv * 192) * 1024ull;

  // Prologue: every wave DMAs its quarter of stage 0 into buffer 0.
  tdm_load_4k(lds_base + (unsigned)wid * 4096u,
              gbase + (unsigned long long)wid * 4096ull);
  __builtin_amdgcn_s_wait_tensorcnt(0);
  __syncthreads();

  v8f c = {};
  v16h a_cur = basisv(xp, grid[0]);      // A for chunk 0, carried across stages
  for (int s = 0; s < NSTAGE; ++s) {
    // Kick off DMA of the next stage into the other buffer (last stage
    // redundantly reloads itself into the idle buffer: branch-free, benign).
    const int sn = (s < NSTAGE - 1) ? s + 1 : NSTAGE - 1;
    tdm_load_4k(lds_base + (unsigned)(((s + 1) & 1) * 16384 + wid * 4096),
                gbase + (unsigned long long)sn * 16384ull +
                    (unsigned long long)wid * 4096ull);

    const int sb = (s & 1) * (STAGE_CHUNKS * CHUNK_HALF) + lane * 16;
    // B pipeline depth 2 (cannot cross the barrier, so restart per stage).
    v16h b_cur = *(const v16h*)&smem[sb];
    v16h b_nxt = *(const v16h*)&smem[sb + CHUNK_HALF];
#pragma unroll
    for (int ci = 0; ci < STAGE_CHUNKS; ++ci) {
      const int g = s * STAGE_CHUNKS + ci;
      // Next A fragment (clamped at the very last chunk; xp is stage-
      // invariant so this legally crosses the stage boundary).
      const int gn = (g + 1 < 192) ? g + 1 : 191;
      v16h a_nxt = basisv(xp, grid[gn]);
      v16h b_fut;
      if (ci < STAGE_CHUNKS - 2)
        b_fut = *(const v16h*)&smem[sb + (ci + 2) * CHUNK_HALF];  // ds_load
      c = __builtin_amdgcn_wmma_f32_16x16x32_f16(
          /*neg_a=*/false, a_cur, /*neg_b=*/false, b_cur,
          /*c_mod=*/(short)0, c, /*reuse_a=*/false, /*reuse_b=*/false);
      a_cur = a_nxt;
      b_cur = b_nxt;
      if (ci < STAGE_CHUNKS - 2) b_nxt = b_fut;
    }

    __builtin_amdgcn_s_wait_tensorcnt(0);  // next-stage DMA landed
    __syncthreads();                       // all waves done reading stage s
  }

  // C/D 16x16 f32 layout: VGPR r -> M = r + 8*grp, N = lane&15
  float* o = part + (size_t)kv * PART_ELEMS +
             (size_t)(n0 + grp * 8) * KO + (lane & 15);
#pragma unroll
  for (int r = 0; r < 8; ++r) o[(size_t)r * KO] = c[r];
}

// ---------------------------------------------------------------------------
// Deterministic K-split reduction (fixed summation order).
// ---------------------------------------------------------------------------
__global__ __launch_bounds__(256) void kan_reduce(
    const float* __restrict__ part, float* __restrict__ out) {
  const int i = blockIdx.x * 256 + threadIdx.x;      // 0 .. 131071 (exact)
  float s = part[i];
#pragma unroll
  for (int p = 1; p < 6; ++p) s += part[(size_t)p * PART_ELEMS + i];
  out[i] = s;
}

extern "C" void kernel_launch(void* const* d_in, const int* in_sizes, int n_in,
                              void* d_out, int out_size, void* d_ws, size_t ws_size,
                              hipStream_t stream) {
  const float* x       = (const float*)d_in[0];   // [8192, 192]
  const float* splines = (const float*)d_in[1];   // [16, 192, 192]
  const float* grid    = (const float*)d_in[2];   // [192]
  float* out           = (float*)d_out;           // [8192, 16]

  // Workspace layout: [0, 1179648)  f16 B-fragment pack (1152 KiB)
  //                   [1179648, +6*131072*4) K-split partials (3 MiB)
  _Float16* bpack = (_Float16*)d_ws;
  float*    part  = (float*)((char*)d_ws + (size_t)NCHUNK * CHUNK_HALF * 2);

  // 1) permute/convert splines -> B fragments (589824 elems)
  kan_pack_splines<<<2304, 256, 0, stream>>>(splines, bpack);
  // 2) WMMA GEMM: 768 blocks = 128 tile-groups x 6 K-splits, 4 waves/block
  kan_wmma_gemm<<<768, 128, 0, stream>>>(x, grid, bpack, part);
  // 3) fixed-order reduction of the 6 partials
  kan_reduce<<<512, 256, 0, stream>>>(part, out);
}
